// TopKAutoEncoder_48962627174599
// MI455X (gfx1250) — matrix operand order repaired and checked by reading, not compiled, not run on hardware
//
#include <hip/hip_runtime.h>

typedef __bf16 bf16_t;
typedef bf16_t bfx4  __attribute__((ext_vector_type(4)));
typedef bf16_t bfx8  __attribute__((ext_vector_type(8)));
typedef bf16_t v16bf __attribute__((ext_vector_type(16)));
typedef float  v8f   __attribute__((ext_vector_type(8)));

#define B_ROWS 4096
#define D_ACT  768
#define D_HID  24576
#define KSEL   32

static __device__ __forceinline__ bf16_t bits_bf16(unsigned short s) {
  union { unsigned short s; bf16_t h; } u; u.s = s; return u.h;
}

// fp32 -> (bf16 hi, bf16 lo) split: hi = truncate-to-bf16(f), lo = bf16(f - hi).
// f - hi is exact in fp32, so hi+lo carries ~16 effective mantissa bits.
static __device__ __forceinline__ void split_f32(float f, bf16_t &hi, bf16_t &lo) {
  union { float f; unsigned u; } a; a.f = f;
  hi = bits_bf16((unsigned short)(a.u >> 16));
  union { unsigned u; float f; } b; b.u = a.u & 0xFFFF0000u;
  float rem = f - b.f;
  union { float f; unsigned u; } c; c.f = rem;
  lo = bits_bf16((unsigned short)(c.u >> 16));
}

static __device__ __forceinline__ v16bf join8(bfx8 a, bfx8 b) {
  return __builtin_shufflevector(a, b, 0,1,2,3,4,5,6,7,8,9,10,11,12,13,14,15);
}

// -------------------------------------------------------------------------
// Kernel 1: acts = (A - b_pre) @ W_enc   [4096 x 768] x [768 x 24576]
// 128x128 tile / workgroup, 8 waves (wave32), bf16 split-3 WMMA.
// -------------------------------------------------------------------------
__global__ __launch_bounds__(256) void enc_gemm_kernel(
    const float* __restrict__ A, const float* __restrict__ W,
    const float* __restrict__ bpre, float* __restrict__ acts)
{
  // Row stride 40 bf16 (80B = 20 banks): conflict-free for 16-lane frag reads.
  __shared__ bf16_t sAhi[128][40];   // [m][k]
  __shared__ bf16_t sAlo[128][40];
  __shared__ bf16_t sBhi[128][40];   // [n][k]  (W_enc tile, transposed)
  __shared__ bf16_t sBlo[128][40];

  const int tid  = threadIdx.x;
  const int lane = tid & 31;
  const int w    = tid >> 5;
  const int wm   = (w & 3) * 32;   // 4 waves along M
  const int wn   = (w >> 2) * 64;  // 2 waves along N
  const int mrow = lane & 15;
  const int kh   = lane >> 4;      // which K-half this lane holds
  const int m0   = blockIdx.y * 128;
  const int n0   = blockIdx.x * 128;

  v8f acc[2][4];
  const v8f vzero = {0.f,0.f,0.f,0.f,0.f,0.f,0.f,0.f};
  #pragma unroll
  for (int i = 0; i < 2; ++i)
    #pragma unroll
    for (int j = 0; j < 4; ++j) acc[i][j] = vzero;

  for (int kt = 0; kt < D_ACT / 32; ++kt) {
    const int k0 = kt * 32;
    __syncthreads();   // protect LDS from previous iteration's readers

    // ---- stage A tile (128x32 fp32), subtract b_pre, split to bf16 hi/lo
    #pragma unroll
    for (int p = 0; p < 4; ++p) {
      const int idx = p * 256 + tid;
      const int m   = idx >> 3;
      const int kq  = (idx & 7) * 4;
      const float4 v  = *(const float4*)(A + (size_t)(m0 + m) * D_ACT + k0 + kq);
      const float4 bp = *(const float4*)(bpre + k0 + kq);
      bf16_t h0,h1,h2,h3,l0,l1,l2,l3;
      split_f32(v.x - bp.x, h0, l0);
      split_f32(v.y - bp.y, h1, l1);
      split_f32(v.z - bp.z, h2, l2);
      split_f32(v.w - bp.w, h3, l3);
      bfx4 hv = {h0,h1,h2,h3};
      bfx4 lv = {l0,l1,l2,l3};
      *(bfx4*)&sAhi[m][kq] = hv;
      *(bfx4*)&sAlo[m][kq] = lv;
    }
    // ---- stage W_enc tile (32x128 fp32), transposed into [n][k]
    #pragma unroll
    for (int p = 0; p < 4; ++p) {
      const int idx = p * 256 + tid;
      const int k   = idx >> 5;
      const int nq  = (idx & 31) * 4;
      const float4 v = *(const float4*)(W + (size_t)(k0 + k) * D_HID + n0 + nq);
      bf16_t h, l;
      split_f32(v.x, h, l); sBhi[nq + 0][k] = h; sBlo[nq + 0][k] = l;
      split_f32(v.y, h, l); sBhi[nq + 1][k] = h; sBlo[nq + 1][k] = l;
      split_f32(v.z, h, l); sBhi[nq + 2][k] = h; sBlo[nq + 2][k] = l;
      split_f32(v.w, h, l); sBhi[nq + 3][k] = h; sBlo[nq + 3][k] = l;
    }
    // prefetch next K-tiles into cache while we compute this one
    if (kt + 1 < D_ACT / 32) {
      const int kn = k0 + 32;
      __builtin_prefetch(A + (size_t)(m0 + (tid >> 1)) * D_ACT + kn + (tid & 1) * 16, 0, 0);
      __builtin_prefetch(W + (size_t)(kn + (tid >> 5)) * D_HID + n0 + (tid & 31) * 4, 0, 0);
    }
    __syncthreads();

    // ---- build fragments per ISA 16-bit A layout:
    // lane<16: K {0..7,16..23}; lane>=16: K {8..15,24..31}
    v16bf aHi[2], aLo[2];
    #pragma unroll
    for (int im = 0; im < 2; ++im) {
      const int m = wm + im * 16 + mrow;
      aHi[im] = join8(*(const bfx8*)&sAhi[m][kh * 8],
                      *(const bfx8*)&sAhi[m][16 + kh * 8]);
      aLo[im] = join8(*(const bfx8*)&sAlo[m][kh * 8],
                      *(const bfx8*)&sAlo[m][16 + kh * 8]);
    }
    // B layout (matches A): lanes 0-15 = K {0..7,16..23} (col=lane), lanes 16-31 = K {8..15,24..31}
    v16bf bHi[4], bLo[4];
    #pragma unroll
    for (int in = 0; in < 4; ++in) {
      const int n = wn + in * 16 + mrow;
      bHi[in] = join8(*(const bfx8*)&sBhi[n][kh * 8],
                      *(const bfx8*)&sBhi[n][16 + kh * 8]);
      bLo[in] = join8(*(const bfx8*)&sBlo[n][kh * 8],
                      *(const bfx8*)&sBlo[n][16 + kh * 8]);
    }
    // ---- split-3 accumulation: hi*hi + lo*hi + hi*lo
    #pragma unroll
    for (int im = 0; im < 2; ++im)
      #pragma unroll
      for (int in = 0; in < 4; ++in) {
        acc[im][in] = __builtin_amdgcn_wmma_f32_16x16x32_bf16(
            false, aHi[im], false, bHi[in], (short)0, acc[im][in], false, false);
        acc[im][in] = __builtin_amdgcn_wmma_f32_16x16x32_bf16(
            false, aLo[im], false, bHi[in], (short)0, acc[im][in], false, false);
        acc[im][in] = __builtin_amdgcn_wmma_f32_16x16x32_bf16(
            false, aHi[im], false, bLo[in], (short)0, acc[im][in], false, false);
      }
  }

  // ---- store C: lane n = lane&15, m = r + 8*(lane>>4)
  #pragma unroll
  for (int im = 0; im < 2; ++im)
    #pragma unroll
    for (int in = 0; in < 4; ++in) {
      const int mgb = m0 + wm + im * 16 + kh * 8;
      const int ng  = n0 + wn + in * 16 + mrow;
      #pragma unroll
      for (int r = 0; r < 8; ++r)
        acts[(size_t)(mgb + r) * D_HID + ng] = acc[im][in][r];
    }
}

// -------------------------------------------------------------------------
// Kernel 2: per-row top-32 (row resident in LDS), emit z, sparse decode.
// -------------------------------------------------------------------------
__global__ __launch_bounds__(256) void topk_dec_kernel(
    const float* __restrict__ acts, const float* __restrict__ Wd,
    const float* __restrict__ bpre, float* __restrict__ z,
    float* __restrict__ arec)
{
  extern __shared__ char smem[];
  float*    srow  = (float*)smem;                    // 24576 fp32 (96 KB)
  unsigned* flags = (unsigned*)(srow + D_HID);       // 768 words (bitmask)
  float*    rv    = (float*)(flags + D_HID / 32);    // 256
  int*      ri    = (int*)(rv + 256);                // 256
  float*    selv  = (float*)(ri + 256);              // 32
  int*      seli  = (int*)(selv + 32);               // 32

  const int tid = threadIdx.x;
  const int row = blockIdx.x;
  const size_t rbase = (size_t)row * D_HID;

  for (int i = tid; i < D_HID; i += 256) srow[i] = acts[rbase + i];
  for (int i = tid; i < D_HID / 32; i += 256) flags[i] = 0u;
  __syncthreads();

  for (int p = 0; p < KSEL; ++p) {
    float best = -__builtin_inff();
    int   bidx = 0x7fffffff;
    for (int i = tid; i < D_HID; i += 256) {
      if ((flags[i >> 5] >> (i & 31)) & 1u) continue;
      const float v = srow[i];
      if (v > best) { best = v; bidx = i; }   // ascending scan => lowest idx on tie
    }
    rv[tid] = best; ri[tid] = bidx;
    __syncthreads();
    for (int s = 128; s > 0; s >>= 1) {
      if (tid < s) {
        const float v2 = rv[tid + s]; const int i2 = ri[tid + s];
        if (v2 > rv[tid] || (v2 == rv[tid] && i2 < ri[tid])) { rv[tid] = v2; ri[tid] = i2; }
      }
      __syncthreads();
    }
    if (tid == 0) {
      selv[p] = rv[0]; seli[p] = ri[0];
      flags[ri[0] >> 5] |= (1u << (ri[0] & 31));   // exclude; srow stays pristine
    }
    __syncthreads();
  }

  // z: selected -> original value, else 0
  for (int i = tid; i < D_HID; i += 256) {
    const float v = ((flags[i >> 5] >> (i & 31)) & 1u) ? srow[i] : 0.0f;
    z[rbase + i] = v;
  }

  // sparse decode: A_rec = b_pre + sum_p selv[p] * W_dec[seli[p], :]
  for (int c = tid; c < D_ACT; c += 256) {
    float a = bpre[c];
    #pragma unroll 8
    for (int p = 0; p < KSEL; ++p)
      a += selv[p] * Wd[(size_t)seli[p] * D_ACT + c];
    arec[(size_t)row * D_ACT + c] = a;
  }
}

// -------------------------------------------------------------------------
extern "C" void kernel_launch(void* const* d_in, const int* in_sizes, int n_in,
                              void* d_out, int out_size, void* d_ws, size_t ws_size,
                              hipStream_t stream) {
  (void)in_sizes; (void)n_in; (void)out_size; (void)d_ws; (void)ws_size;
  const float* A     = (const float*)d_in[0];
  const float* W_enc = (const float*)d_in[1];
  const float* W_dec = (const float*)d_in[2];
  const float* b_pre = (const float*)d_in[3];
  // outputs concatenated flat: (A_reconstruct, acts, z)
  float* arec = (float*)d_out;
  float* acts = arec + (size_t)B_ROWS * D_ACT;
  float* z    = acts + (size_t)B_ROWS * D_HID;

  dim3 g1(D_HID / 128, B_ROWS / 128);
  enc_gemm_kernel<<<g1, 256, 0, stream>>>(A, W_enc, b_pre, acts);

  const size_t sm2 = (size_t)D_HID * 4 + (D_HID / 32) * 4 + 256 * 4 + 256 * 4
                   + 32 * 4 + 32 * 4;
  topk_dec_kernel<<<B_ROWS, 256, sm2, stream>>>(acts, W_dec, b_pre, z, arec);
}
